// LSTMDecoder_11716670784123
// MI455X (gfx1250) — compile-verified, matrix operand order checked by reading, not run on hardware
//
#include <hip/hip_runtime.h>
#include <hip/hip_bf16.h>
#include <stdint.h>

// LSTM decoder with Luong-concat attention for MI455X (gfx1250, wave32).
// All matmuls use v_wmma_f32_16x16x32_f16 (fp16 in, fp32 accumulate).
// en_proj is hoisted out of the time loop; past_attn is dead code (dropped).
// Each wave computes TWO 16x16 output tiles with a shared A fragment and two
// independent accumulators -> two independent WMMA dependency chains (ILP).

typedef _Float16 f16;
typedef _Float16 v8h  __attribute__((ext_vector_type(8)));
typedef _Float16 v16h __attribute__((ext_vector_type(16)));
typedef float    v8f  __attribute__((ext_vector_type(8)));

#define B_ 16
#define T_ 64
#define S_ 256
#define I_ 512
#define H_ 512

// ---------------------------------------------------------------------------
// WMMA fragment load: row-major source, K-contiguous.
// lane&15 selects the row; (lane>>4) selects which 8-half K-group.
// Elements 0..7  = K {k..k+7},   elements 8..15 = K {k+16..k+23}.
// ---------------------------------------------------------------------------
__device__ __forceinline__ v16h ldfrag(const f16* __restrict__ rowp, int k) {
    v8h lo = *(const v8h*)(rowp + k);
    v8h hi = *(const v8h*)(rowp + k + 16);
    v16h r;
#pragma unroll
    for (int i = 0; i < 8; ++i) { r[i] = lo[i]; r[i + 8] = hi[i]; }
    return r;
}

// Generic fp16 GEMM: D[M,N] = A[M,K] * B[N,K]^T  (i.e. X @ W.T), fp32 out.
// Two adjacent 16x16 N-tiles per wave (Ntiles must be even), grid-stride over
// tile pairs. EXEC stays all-1s (wave-uniform control flow), as WMMA requires.
__global__ void __launch_bounds__(256)
wmma_gemm_f16(const f16* __restrict__ A, int lda,
              const f16* __restrict__ Bw, int ldb,
              float* __restrict__ D, int ldd,
              int Mtiles, int Ntiles, int K) {
    const int lane  = threadIdx.x & 31;
    const int wave  = blockIdx.x * (blockDim.x >> 5) + (threadIdx.x >> 5);
    const int nwave = gridDim.x * (blockDim.x >> 5);
    const int row   = lane & 15;          // A row / B row (== D column) in tile
    const int khalf = (lane >> 4) << 3;   // 0 or 8
    const int mbase = (lane >> 4) << 3;   // D: lanes 16..31 hold M = 8..15
    const int npairs = Ntiles >> 1;
    const int total  = Mtiles * npairs;

    for (int pair = wave; pair < total; pair += nwave) {
        const int mt  = pair / npairs;
        const int np  = pair - mt * npairs;
        const int nt0 = np * 2;
        const f16* arow  = A  + (size_t)(mt * 16 + row) * lda;
        const f16* brow0 = Bw + (size_t)(nt0 * 16 + row) * ldb;
        const f16* brow1 = brow0 + (size_t)16 * ldb;
        v8f acc0 = {}, acc1 = {};
        for (int k0 = 0; k0 < K; k0 += 32) {
            v16h af  = ldfrag(arow,  k0 + khalf);
            v16h bf0 = ldfrag(brow0, k0 + khalf);
            v16h bf1 = ldfrag(brow1, k0 + khalf);
            acc0 = __builtin_amdgcn_wmma_f32_16x16x32_f16(
                false, af, false, bf0, (short)0, acc0, false, false);
            acc1 = __builtin_amdgcn_wmma_f32_16x16x32_f16(
                false, af, false, bf1, (short)0, acc1, false, false);
        }
        float* dp0 = D + (size_t)(mt * 16 + mbase) * ldd + nt0 * 16 + row;
#pragma unroll
        for (int v = 0; v < 8; ++v) {
            dp0[(size_t)v * ldd]      = acc0[v];
            dp0[(size_t)v * ldd + 16] = acc1[v];
        }
    }
}

// ---------------------------------------------------------------------------
// Setup kernels (run once per launch)
// ---------------------------------------------------------------------------
__global__ void cvt2d_kernel(const float* __restrict__ src, int sld,
                             f16* __restrict__ dst, int dld, int R, int C) {
    int n = R * C;
    for (int idx = blockIdx.x * blockDim.x + threadIdx.x; idx < n;
         idx += gridDim.x * blockDim.x) {
        int r = idx / C, c = idx - r * C;
        dst[(size_t)r * dld + c] = (f16)src[(size_t)r * sld + c];
    }
}

// encoder_hy [S,B,H] -> enhy16 [B*S, H] (b-major rows), fp16
__global__ void cvt_enhy_kernel(const float* __restrict__ enc,
                                f16* __restrict__ dst) {
    int n = B_ * S_ * H_;
    for (int idx = blockIdx.x * blockDim.x + threadIdx.x; idx < n;
         idx += gridDim.x * blockDim.x) {
        int r = idx >> 9, h = idx & (H_ - 1);
        int b = r >> 8, s = r & (S_ - 1);
        dst[idx] = (f16)enc[((size_t)s * B_ + b) * H_ + h];
    }
}

// xin16 row layout per batch b: [ x_t (512) | h_attn (512) | h (512) ]
__global__ void init_state_kernel(const float* __restrict__ h0,
                                  const float* __restrict__ c0,
                                  const float* __restrict__ ha0,
                                  float* __restrict__ c,
                                  f16* __restrict__ xin16,
                                  f16* __restrict__ cat16) {
    int idx = blockIdx.x * blockDim.x + threadIdx.x;
    if (idx >= B_ * H_) return;
    int b = idx >> 9, j = idx & (H_ - 1);
    c[idx] = c0[idx];
    xin16[b * 1536 + 1024 + j] = (f16)h0[idx];
    xin16[b * 1536 +  512 + j] = (f16)ha0[idx];
    cat16[b * 1024 +  512 + j] = (f16)h0[idx];
}

// ---------------------------------------------------------------------------
// Per-step kernels
// ---------------------------------------------------------------------------
__global__ void pack_x_kernel(const float* __restrict__ input_,
                              f16* __restrict__ xin16, int t) {
    int idx = blockIdx.x * blockDim.x + threadIdx.x;
    if (idx >= B_ * I_) return;
    int b = idx >> 9, j = idx & (I_ - 1);
    xin16[b * 1536 + j] = (f16)input_[((size_t)b * T_ + t) * I_ + j];
}

__device__ __forceinline__ float sigm(float x) {
    return 1.0f / (1.0f + __expf(-x));
}

__global__ void lstm_cell_kernel(const float* __restrict__ gates,
                                 const float* __restrict__ b_ih,
                                 const float* __restrict__ b_hh,
                                 float* __restrict__ c,
                                 f16* __restrict__ xin16,
                                 f16* __restrict__ cat16) {
    int idx = blockIdx.x * blockDim.x + threadIdx.x;
    if (idx >= B_ * H_) return;
    int b = idx >> 9, j = idx & (H_ - 1);
    const float* g = gates + b * (4 * H_);
    float gi = g[j]            + b_ih[j]            + b_hh[j];
    float gf = g[H_ + j]       + b_ih[H_ + j]       + b_hh[H_ + j];
    float gg = g[2 * H_ + j]   + b_ih[2 * H_ + j]   + b_hh[2 * H_ + j];
    float go = g[3 * H_ + j]   + b_ih[3 * H_ + j]   + b_hh[3 * H_ + j];
    float c2 = sigm(gf) * c[idx] + sigm(gi) * tanhf(gg);
    float h2 = sigm(go) * tanhf(c2);
    c[idx] = c2;
    xin16[b * 1536 + 1024 + j] = (f16)h2;   // h slice of next-step xin
    cat16[b * 1024 +  512 + j] = (f16)h2;   // h slice of [ctx ; h]
}

// scores[b,s] = W_warp . tanh(en_proj[b,s,:] + de[b,:]) — one wave per row.
__global__ void __launch_bounds__(256)
scores_kernel(const float* __restrict__ en_proj,
              const float* __restrict__ de,
              const float* __restrict__ W_warp,
              float* __restrict__ scores) {
    int lane = threadIdx.x & 31;
    int gw = blockIdx.x * (blockDim.x >> 5) + (threadIdx.x >> 5);
    if (gw >= B_ * S_) return;
    int b = gw >> 8;
    const float* ep = en_proj + (size_t)gw * H_;
    const float* dp = de + b * H_;
    float acc = 0.f;
    for (int h = lane; h < H_; h += 32)
        acc = fmaf(tanhf(ep[h] + dp[h]), W_warp[h], acc);
#pragma unroll
    for (int off = 16; off > 0; off >>= 1) acc += __shfl_xor(acc, off, 32);
    if (lane == 0) scores[gw] = acc;
}

// Per-batch softmax over S, then ctx[b,h] = sum_s attn * enhy[b,s,h] (fp32).
__global__ void softmax_ctx_kernel(const float* __restrict__ scores,
                                   const float* __restrict__ enc, // [S,B,H]
                                   f16* __restrict__ cat16) {
    __shared__ float red[S_];
    __shared__ float attn[S_];
    int b = blockIdx.x, tid = threadIdx.x;
    float v = scores[b * S_ + tid];
    red[tid] = v; __syncthreads();
    for (int o = 128; o > 0; o >>= 1) {
        if (tid < o) red[tid] = fmaxf(red[tid], red[tid + o]);
        __syncthreads();
    }
    float m = red[0]; __syncthreads();
    float e = __expf(v - m);
    red[tid] = e; __syncthreads();
    for (int o = 128; o > 0; o >>= 1) {
        if (tid < o) red[tid] += red[tid + o];
        __syncthreads();
    }
    attn[tid] = e / red[0];
    __syncthreads();
    float a0 = 0.f, a1 = 0.f;
    for (int s = 0; s < S_; ++s) {
        float w = attn[s];
        const float* rowp = enc + ((size_t)s * B_ + b) * H_;
        a0 = fmaf(w, rowp[tid], a0);          // coalesced over h
        a1 = fmaf(w, rowp[tid + 256], a1);
    }
    cat16[b * 1024 + tid]       = (f16)a0;    // ctx slice of [ctx ; h]
    cat16[b * 1024 + tid + 256] = (f16)a1;
}

__global__ void out_epilogue_kernel(const float* __restrict__ pre,
                                    float* __restrict__ out,
                                    f16* __restrict__ xin16, int t) {
    int idx = blockIdx.x * blockDim.x + threadIdx.x;
    if (idx >= B_ * H_) return;
    int b = idx >> 9, j = idx & (H_ - 1);
    float v = tanhf(pre[idx]);
    out[((size_t)b * T_ + t) * H_ + j] = v;       // d_out is [B,T,H]
    xin16[b * 1536 + 512 + j] = (f16)v;           // h_attn for next step
}

// ---------------------------------------------------------------------------
// Launcher
// ---------------------------------------------------------------------------
extern "C" void kernel_launch(void* const* d_in, const int* in_sizes, int n_in,
                              void* d_out, int out_size, void* d_ws, size_t ws_size,
                              hipStream_t stream) {
    const float* input_ = (const float*)d_in[0];
    const float* h0     = (const float*)d_in[1];
    const float* c0     = (const float*)d_in[2];
    const float* ha0    = (const float*)d_in[3];
    const float* enc    = (const float*)d_in[4];   // [S,B,H]
    // d_in[5] hidden_attn, d_in[6] past_attn0: unused (dead in reference output)
    const float* W_ih   = (const float*)d_in[7];   // [2048,1024]
    const float* b_ih   = (const float*)d_in[8];
    const float* W_hh   = (const float*)d_in[9];   // [2048,512]
    const float* b_hh   = (const float*)d_in[10];
    const float* W_en   = (const float*)d_in[11];  // [512,512]
    const float* W_de   = (const float*)d_in[12];  // [512,512]
    const float* W_warp = (const float*)d_in[13];  // [1,512]
    const float* W_out  = (const float*)d_in[14];  // [512,1024]
    float* out = (float*)d_out;

    // Workspace carve-up (~21.5 MB total)
    uint8_t* p = (uint8_t*)d_ws;
    auto alloc = [&](size_t bytes) -> void* {
        void* r = (void*)p;
        p += (bytes + 255) & ~(size_t)255;
        return r;
    };
    f16*   Wg16    = (f16*)  alloc((size_t)2048 * 1536 * 2); // [W_ih | W_hh]
    f16*   Wde16   = (f16*)  alloc((size_t)512 * 512 * 2);
    f16*   Wout16  = (f16*)  alloc((size_t)512 * 1024 * 2);
    f16*   Wen16   = (f16*)  alloc((size_t)512 * 512 * 2);
    f16*   enhy16  = (f16*)  alloc((size_t)B_ * S_ * H_ * 2);
    float* en_proj = (float*)alloc((size_t)B_ * S_ * H_ * 4);
    float* gates   = (float*)alloc((size_t)B_ * 4 * H_ * 4);
    float* de      = (float*)alloc((size_t)B_ * H_ * 4);
    float* scores  = (float*)alloc((size_t)B_ * S_ * 4);
    float* cst     = (float*)alloc((size_t)B_ * H_ * 4);
    float* pre_out = (float*)alloc((size_t)B_ * H_ * 4);
    f16*   xin16   = (f16*)  alloc((size_t)B_ * 1536 * 2);   // [x|h_attn|h]
    f16*   cat16   = (f16*)  alloc((size_t)B_ * 1024 * 2);   // [ctx|h]
    (void)ws_size; (void)in_sizes; (void)n_in; (void)out_size;

    // --- one-time setup ---
    cvt2d_kernel<<<256, 256, 0, stream>>>(W_ih, 1024, Wg16,        1536, 2048, 1024);
    cvt2d_kernel<<<256, 256, 0, stream>>>(W_hh,  512, Wg16 + 1024, 1536, 2048,  512);
    cvt2d_kernel<<<128, 256, 0, stream>>>(W_de,  512, Wde16,        512,  512,  512);
    cvt2d_kernel<<<128, 256, 0, stream>>>(W_out, 1024, Wout16,     1024,  512, 1024);
    cvt2d_kernel<<<128, 256, 0, stream>>>(W_en,  512, Wen16,        512,  512,  512);
    cvt_enhy_kernel<<<512, 256, 0, stream>>>(enc, enhy16);
    init_state_kernel<<<32, 256, 0, stream>>>(h0, c0, ha0, cst, xin16, cat16);

    // en_proj[b,s,:] = enhy[b,s,:] @ W_en.T : M=4096, N=512, K=512
    // 256 M-tiles x 16 N-tile-pairs = 4096 wave-jobs
    wmma_gemm_f16<<<512, 256, 0, stream>>>(enhy16, 512, Wen16, 512,
                                           en_proj, 512, 256, 32, 512);

    // --- sequential decode ---
    for (int t = 0; t < T_; ++t) {
        pack_x_kernel<<<32, 256, 0, stream>>>(input_, xin16, t);
        // gates = [x|h_attn|h] @ [W_ih|W_hh].T : M=16, N=2048, K=1536 (64 pairs)
        wmma_gemm_f16<<<8, 256, 0, stream>>>(xin16, 1536, Wg16, 1536,
                                             gates, 2048, 1, 128, 1536);
        lstm_cell_kernel<<<32, 256, 0, stream>>>(gates, b_ih, b_hh, cst,
                                                 xin16, cat16);
        // de = h2 @ W_de.T : M=16, N=512, K=512 (16 pairs)
        wmma_gemm_f16<<<2, 256, 0, stream>>>(xin16 + 1024, 1536, Wde16, 512,
                                             de, 512, 1, 32, 512);
        scores_kernel<<<512, 256, 0, stream>>>(en_proj, de, W_warp, scores);
        softmax_ctx_kernel<<<B_, 256, 0, stream>>>(scores, enc, cat16);
        // pre_out = [ctx|h2] @ W_out.T : M=16, N=512, K=1024 (16 pairs)
        wmma_gemm_f16<<<2, 256, 0, stream>>>(cat16, 1024, Wout16, 1024,
                                             pre_out, 512, 1, 32, 1024);
        out_epilogue_kernel<<<32, 256, 0, stream>>>(pre_out, out, xin16, t);
    }
}
